// Conv2d_mod_32083405701533
// MI455X (gfx1250) — matrix operand order repaired and checked by reading, not compiled, not run on hardware
//
#include <hip/hip_runtime.h>
#include <hip/hip_bf16.h>
#include <math.h>

// ---------------- problem constants ----------------
#define B_SZ   16
#define C_IN   512
#define C_OUT  512
#define LAT    256
#define HH     32
#define WW     32
#define HW     (HH*WW)        // 1024
#define KTOT   (C_IN*9)       // 4608, ordered (pos, channel): k = pos*512 + i
#define NIT    144            // 9 positions * 16 k-steps of 32

#define EQ_CONV (1.0f/48.0f)        // sqrt(2/(512*9))
#define EQ_LIN  (0.08838834765f)    // sqrt(2/256)

// ---------------- vector types ----------------
typedef __attribute__((ext_vector_type(16))) __bf16       v16bf;
typedef __attribute__((ext_vector_type(8)))  __bf16       v8bf;
typedef __attribute__((ext_vector_type(8)))  float        v8f;
typedef __attribute__((ext_vector_type(4)))  unsigned int u32x4;
typedef __attribute__((ext_vector_type(8)))  int          i32x8;
typedef __attribute__((ext_vector_type(4)))  int          i32x4;

__device__ __forceinline__ unsigned short f2bf(float f) {
  unsigned u = __builtin_bit_cast(unsigned, f);
  unsigned r = u + 0x7FFFu + ((u >> 16) & 1u);   // round-to-nearest-even
  return (unsigned short)(r >> 16);
}

// ---------------- prologue 1: repack weights to bf16, (pos,channel) K-order --
__global__ __launch_bounds__(256) void convert_w_kernel(
    const float* __restrict__ w, unsigned short* __restrict__ wq) {
  int idx = blockIdx.x * 256 + threadIdx.x;
  const int total = C_OUT * KTOT;
  for (; idx < total; idx += gridDim.x * 256) {
    int o   = idx / KTOT;
    int rem = idx - o * KTOT;
    int pos = rem >> 9;          // 0..8
    int i   = rem & 511;         // channel
    int kh  = pos / 3, kw = pos % 3;
    float v = w[(((size_t)o * C_IN + i) * 3 + kh) * 3 + kw] * EQ_CONV;
    wq[idx] = f2bf(v);
  }
}

// ---------------- prologue 2: sum of squared (equalized) weights per o -------
__global__ __launch_bounds__(256) void sumsq_kernel(
    const float* __restrict__ w, float* __restrict__ sumsq) {
  __shared__ float red[256];
  const int o = blockIdx.x;
  const float* row = w + (size_t)o * KTOT;
  float acc = 0.f;
  for (int j = threadIdx.x; j < KTOT; j += 256) {
    float v = row[j] * EQ_CONV;
    acc += v * v;
  }
  red[threadIdx.x] = acc;
  __syncthreads();
  for (int s = 128; s > 0; s >>= 1) {
    if (threadIdx.x < s) red[threadIdx.x] += red[threadIdx.x + s];
    __syncthreads();
  }
  if (threadIdx.x == 0) sumsq[o] = red[0];
}

// ---------------- prologue 3: style GEMM + demod -> per (b,o) scale ----------
__global__ __launch_bounds__(256) void style_kernel(
    const float* __restrict__ style_w, const float* __restrict__ lin_w,
    const float* __restrict__ lin_b,   const float* __restrict__ mod_bias,
    const float* __restrict__ sumsq,   float* __restrict__ scale) {
  int idx = blockIdx.x * 256 + threadIdx.x;
  if (idx >= B_SZ * C_OUT) return;
  int bb = idx >> 9;
  int o  = idx & 511;
  const float* sw = style_w + (size_t)bb * LAT;
  const float* lw = lin_w   + (size_t)o  * LAT;
  float acc = 0.f;
  #pragma unroll 4
  for (int j = 0; j < LAT; ++j) acc += sw[j] * lw[j];
  float s = acc * EQ_LIN + lin_b[o] + 1.0f + mod_bias[o];
  float d = rsqrtf(s * s * sumsq[o] + 1e-8f);
  scale[idx] = s * d;
}

// ---------------- main: 16x per-sample implicit GEMM via bf16 WMMA -----------
// Block tile 256(M=O) x 128(N=HW), K-step 32; 8 waves 4(M)x2(N), wave tile
// 64x64 = 16 WMMA/step. A-tile (weights, dense 256x32 bf16 rectangle) is DMA'd
// global->LDS by the Tensor Data Mover (one tensor_load_to_lds per step,
// TENSORcnt-tracked, fully async under the WMMAs). B-tile (im2col with halo
// zero-fill, not TDM-expressible) is software-pipelined through registers.
// LDS double-buffered, one barrier per K-step.
__global__ __launch_bounds__(256) void modconv_wmma(
    const float* __restrict__ x, const unsigned short* __restrict__ wq,
    const float* __restrict__ scale, const float* __restrict__ bias,
    float* __restrict__ out) {
  __shared__ unsigned short As[2][256][32];   // [buf][o_local][k]  32 KB
  __shared__ unsigned short Bs[2][128][32];   // [buf][px_local][k] 16 KB

  const int bb   = blockIdx.z;
  const int m0   = blockIdx.y * 256;       // output-channel tile
  const int n0   = blockIdx.x * 128;       // pixel tile
  const int tid  = threadIdx.x;
  const int lane = tid & 31;
  const int wave = tid >> 5;
  const int wm   = (wave >> 1) * 64;       // wave M offset (0..192)
  const int wn   = (wave & 1) * 64;        // wave N offset (0 / 64)

  v8f acc[4][4] = {};

  // ---- TDM descriptor (D#) for the A tile: 256 rows x 32 bf16 ----
  // group1 is constant: data_size=2B(enc 1), tensor 4608x512, tile 32x256,
  // dim0 stride 4608 elements.
  const unsigned long long abase =
      (unsigned long long)(size_t)wq + (unsigned long long)m0 * (KTOT * 2);
  const unsigned lds_a0 = (unsigned)(size_t)&As[0][0][0];
  i32x8 g1d;
  g1d[0] = (int)(1u << 16);                                   // data_size=1 -> 2B
  g1d[1] = (int)((KTOT & 0xFFFFu) << 16);                     // tensor_dim0 lo16
  g1d[2] = (int)(((KTOT >> 16) & 0xFFFFu) | ((C_OUT & 0xFFFFu) << 16)); // dim0 hi | dim1 lo
  g1d[3] = (int)(((C_OUT >> 16) & 0xFFFFu) | (32u << 16));    // dim1 hi | tile_dim0=32
  g1d[4] = (int)256;                                          // tile_dim1=256, tile_dim2=0
  g1d[5] = (int)KTOT;                                         // tensor_dim0_stride lo32
  g1d[6] = 0;
  g1d[7] = 0;
  const i32x4 gz  = {0, 0, 0, 0};
#if __clang_major__ >= 23
  const i32x8 gz8 = {0, 0, 0, 0, 0, 0, 0, 0};
#endif

  auto issueA = [&](int it, int buf) {              // wave-uniform call
    const unsigned long long ga =
        abase + (unsigned long long)((it >> 4) * C_IN + (it & 15) * 32) * 2u;
    u32x4 g0;
    g0[0] = 1u;                                     // count=1, user mode
    g0[1] = lds_a0 + (unsigned)buf * (unsigned)sizeof(As[0]);   // lds_addr
    g0[2] = (unsigned)ga;                           // global_addr[31:0]
    g0[3] = (unsigned)(ga >> 32) | (2u << 30);      // global_addr[56:32] | type=2
#if __clang_major__ >= 23
    __builtin_amdgcn_tensor_load_to_lds(g0, g1d, gz, gz, gz8, 0);
#else
    __builtin_amdgcn_tensor_load_to_lds(g0, g1d, gz, gz, 0);
#endif
  };

  // ---- B-stage: thread -> (pixel, 16-channel half), regs pipeline ----
  const int b_px = tid >> 1;
  const int b_kh = tid & 1;
  const int npx  = n0 + b_px;
  const int ph   = npx >> 5, pw = npx & 31;
  const float* xb = x + (size_t)bb * C_IN * HW;

  unsigned int breg[8];
  auto stageB = [&](int it) {
    const int pos = it >> 4;
    const int ks  = it & 15;
    const int hs  = ph + pos / 3 - 1;
    const int wsx = pw + pos % 3 - 1;
    const bool valid = ((unsigned)hs < (unsigned)HH) & ((unsigned)wsx < (unsigned)WW);
    const float* xsrc = xb + hs * WW + wsx + (size_t)(ks * 32 + b_kh * 16) * HW;
    if (valid) __builtin_prefetch(xsrc + 32 * HW, 0, 0);   // global_prefetch_b8
    #pragma unroll
    for (int j2 = 0; j2 < 8; ++j2) {
      unsigned lo = valid ? (unsigned)f2bf(xsrc[(2 * j2 + 0) * HW]) : 0u;
      unsigned hi = valid ? (unsigned)f2bf(xsrc[(2 * j2 + 1) * HW]) : 0u;
      breg[j2] = lo | (hi << 16);
    }
  };
  auto drainB = [&](int buf) {
    unsigned int* bd = (unsigned int*)&Bs[buf][b_px][b_kh * 16];
    #pragma unroll
    for (int j2 = 0; j2 < 8; ++j2) bd[j2] = breg[j2];
  };

  // per-lane fragment gather parameters (contiguous k order: half-wave h holds k = 16h..16h+15 for A and B)
  const int frow  = lane & 15;
  const int abse  = lane & 16;             // A: K base 0 / 16
  const int bbse  = lane & 16;             // B: K base 0 / 16

  // ---- prologue of the pipeline ----
  if (wave == 0) issueA(0, 0);
  stageB(0);
  drainB(0);
  if (wave == 0) __builtin_amdgcn_s_wait_tensorcnt(0);
  __syncthreads();

  #pragma unroll 1
  for (int it = 0; it < NIT; ++it) {
    const int buf = it & 1;
    if (wave == 0 && it + 1 < NIT) issueA(it + 1, buf ^ 1);  // async A DMA
    if (it + 1 < NIT) stageB(it + 1);                        // B global->regs

    // ---- fragment loads from current buffer ----
    v16bf af[4], bfr[4];
    #pragma unroll
    for (int i = 0; i < 4; ++i) {
      const __bf16* ap = (const __bf16*)&As[buf][wm + i * 16 + frow][0];
      v8bf lo = *(const v8bf*)(ap + abse);           // K = abse..abse+7
      v8bf hi = *(const v8bf*)(ap + abse + 8);       // K = abse+8..+15
      af[i] = __builtin_shufflevector(lo, hi, 0,1,2,3,4,5,6,7,8,9,10,11,12,13,14,15);
    }
    #pragma unroll
    for (int j = 0; j < 4; ++j) {
      const __bf16* bp = (const __bf16*)&Bs[buf][wn + j * 16 + frow][0];
      v8bf lo = *(const v8bf*)(bp + bbse);           // K = bbse..bbse+7
      v8bf hi = *(const v8bf*)(bp + bbse + 8);       // K = bbse+8..+15
      bfr[j] = __builtin_shufflevector(lo, hi, 0,1,2,3,4,5,6,7,8,9,10,11,12,13,14,15);
    }

    // ---- 16 WMMAs per wave per K-step ----
    #pragma unroll
    for (int i = 0; i < 4; ++i)
      #pragma unroll
      for (int j = 0; j < 4; ++j)
        acc[i][j] = __builtin_amdgcn_wmma_f32_16x16x32_bf16(
            false, af[i], false, bfr[j], (short)0, acc[i][j], false, false);

    if (it + 1 < NIT) drainB(buf ^ 1);
    if (wave == 0 && it + 1 < NIT) __builtin_amdgcn_s_wait_tensorcnt(0);
    __syncthreads();
  }

  // ---- epilogue: fold modulation/demodulation scale + bias ----
  const int col = lane & 15;
  const int rhi = (lane & 16) >> 1;   // +8 rows for upper half-wave
  #pragma unroll
  for (int i = 0; i < 4; ++i) {
    #pragma unroll
    for (int r = 0; r < 8; ++r) {
      const int m = m0 + wm + i * 16 + r + rhi;
      const float sc = scale[bb * C_OUT + m];
      const float bi = bias[m];
      float* orow = out + ((size_t)bb * C_OUT + m) * HW + n0;
      #pragma unroll
      for (int j = 0; j < 4; ++j)
        orow[wn + j * 16 + col] = acc[i][j][r] * sc + bi;
    }
  }
}

// ---------------- launch ----------------
extern "C" void kernel_launch(void* const* d_in, const int* in_sizes, int n_in,
                              void* d_out, int out_size, void* d_ws, size_t ws_size,
                              hipStream_t stream) {
  const float* x        = (const float*)d_in[0];
  const float* style_w  = (const float*)d_in[1];
  const float* conv_w   = (const float*)d_in[2];
  const float* lin_w    = (const float*)d_in[3];
  const float* lin_b    = (const float*)d_in[4];
  const float* mod_bias = (const float*)d_in[5];
  const float* conv_b   = (const float*)d_in[6];
  float* out = (float*)d_out;

  char* ws = (char*)d_ws;
  unsigned short* wq = (unsigned short*)ws;                       // 4,718,592 B
  float* sumsq = (float*)(ws + (size_t)C_OUT * KTOT * 2);         // 512 f32
  float* scl   = sumsq + C_OUT;                                   // 16*512 f32

  convert_w_kernel<<<2304, 256, 0, stream>>>(conv_w, wq);
  sumsq_kernel<<<C_OUT, 256, 0, stream>>>(conv_w, sumsq);
  style_kernel<<<(B_SZ * C_OUT) / 256, 256, 0, stream>>>(
      style_w, lin_w, lin_b, mod_bias, sumsq, scl);

  dim3 grid(HW / 128, C_OUT / 256, B_SZ);
  modconv_wmma<<<grid, 256, 0, stream>>>(x, wq, scl, conv_b, out);
}
